// OuterProductNetworkLayer_83167746719776
// MI455X (gfx1250) — matrix operand order repaired and checked by reading, not compiled, not run on hardware
//
#include <hip/hip_runtime.h>

#define NF  32
#define EMB 64
#define PP  496
#define BB  2048
#define LDS_STRIDE 72   // padded halfs per e-row: (72*l)%128 distinct for 16 lanes

typedef __attribute__((ext_vector_type(16))) _Float16 v16h;
typedef __attribute__((ext_vector_type(8)))  _Float16 v8h;
typedef __attribute__((ext_vector_type(8)))  float    v8f;

__global__ __launch_bounds__(256)
void opnl_wmma_kernel(const float* __restrict__ emb,
                      const float* __restrict__ kern,
                      float* __restrict__ out)
{
    __shared__ __attribute__((aligned(16))) _Float16 Kt[EMB * LDS_STRIDE + 8];

    const int n    = blockIdx.x;   // pair index 0..495
    const int bblk = blockIdx.y;   // batch block of 128 rows

    // Decode upper-triangle pair n -> (row_f, col_f), wave-uniform scalar loop.
    int r = 0, rem = n;
    while (rem >= (NF - 1 - r)) { rem -= (NF - 1 - r); ++r; }
    const int row_f = r;            // ROW_IDX[n]  (p side)
    const int col_f = r + 1 + rem;  // COL_IDX[n]  (q side)

    // ---- Stage K_n (64x64, [h][e]) into LDS as f16, e-major transposed.
    {
        const int t  = threadIdx.x;       // 0..255
        const int h  = t >> 2;            // 0..63
        const int e0 = (t & 3) * 16;      // 0,16,32,48
        const float4* src =
            (const float4*)(kern + ((size_t)h * PP + n) * EMB + e0);
#pragma unroll
        for (int j = 0; j < 4; ++j) {
            float4 v = src[j];
            const int e = e0 + j * 4;
            Kt[(e + 0) * LDS_STRIDE + h] = (_Float16)v.x;
            Kt[(e + 1) * LDS_STRIDE + h] = (_Float16)v.y;
            Kt[(e + 2) * LDS_STRIDE + h] = (_Float16)v.z;
            Kt[(e + 3) * LDS_STRIDE + h] = (_Float16)v.w;
        }
    }
    __syncthreads();

    const int lane = threadIdx.x & 31;
    const int wave = threadIdx.x >> 5;    // 0..7
    const int l16  = lane & 15;
    const int g    = lane >> 4;           // lane half
    const int brow0 = bblk * 128 + wave * 16;

    // ---- A fragments: Q rows (field col_f). ISA 16-bit A layout:
    //      halfs 0..7  = K[32*kk + 8g .. +7], halfs 8..15 = K[32*kk + 16 + 8g .. +7]
    const float* qrow = emb + ((size_t)(brow0 + l16) * NF + col_f) * EMB;
    v16h a[2];
#pragma unroll
    for (int kk = 0; kk < 2; ++kk) {
        const int k0 = 32 * kk + 8 * g;
        float4 t0 = *(const float4*)(qrow + k0);
        float4 t1 = *(const float4*)(qrow + k0 + 4);
        float4 t2 = *(const float4*)(qrow + k0 + 16);
        float4 t3 = *(const float4*)(qrow + k0 + 20);
        a[kk][0]  = (_Float16)t0.x; a[kk][1]  = (_Float16)t0.y;
        a[kk][2]  = (_Float16)t0.z; a[kk][3]  = (_Float16)t0.w;
        a[kk][4]  = (_Float16)t1.x; a[kk][5]  = (_Float16)t1.y;
        a[kk][6]  = (_Float16)t1.z; a[kk][7]  = (_Float16)t1.w;
        a[kk][8]  = (_Float16)t2.x; a[kk][9]  = (_Float16)t2.y;
        a[kk][10] = (_Float16)t2.z; a[kk][11] = (_Float16)t2.w;
        a[kk][12] = (_Float16)t3.x; a[kk][13] = (_Float16)t3.y;
        a[kk][14] = (_Float16)t3.z; a[kk][15] = (_Float16)t3.w;
    }

    // ---- T(16x64) = Q(16x64) @ K_n(64x64): 4 column blocks x 2 K-chunks.
    //      B layout: lane N = l16, halfs 0..7 = K[8g..+7], halfs 8..15 = K[16+8g..+7].
    v8f acc[4] = {v8f{0.f}, v8f{0.f}, v8f{0.f}, v8f{0.f}};
#pragma unroll
    for (int kk = 0; kk < 2; ++kk) {
#pragma unroll
        for (int c = 0; c < 4; ++c) {
            const int e = c * 16 + l16;
            const _Float16* bp = &Kt[e * LDS_STRIDE + 32 * kk + 8 * g];
            v8h b0 = *(const v8h*)bp;
            v8h b1 = *(const v8h*)(bp + 16);
            v16h b;
#pragma unroll
            for (int i = 0; i < 8; ++i) { b[i] = b0[i]; b[8 + i] = b1[i]; }
            acc[c] = __builtin_amdgcn_wmma_f32_16x16x32_f16(
                false, a[kk], false, b, (short)0, acc[c], false, false);
        }
    }

    // ---- Epilogue: out[m] = sum_e T[m,e] * P[m,e].
    //      C layout: lane (l16, g), vgpr v -> row m = v + 8g, col e = 16c + l16.
    float sums[8];
#pragma unroll
    for (int v = 0; v < 8; ++v) {
        const int m = 8 * g + v;
        const float* prow =
            emb + ((size_t)(brow0 + m) * NF + row_f) * EMB + l16;
        float s = acc[0][v] * prow[0]
                + acc[1][v] * prow[16]
                + acc[2][v] * prow[32]
                + acc[3][v] * prow[48];
#pragma unroll
        for (int off = 8; off >= 1; off >>= 1)
            s += __shfl_xor(s, off, 32);
        sums[v] = s;
    }
    if (l16 == 0) {
#pragma unroll
        for (int v = 0; v < 8; ++v) {
            const int m = 8 * g + v;
            out[(size_t)(brow0 + m) * PP + n] = sums[v];
        }
    }
}

extern "C" void kernel_launch(void* const* d_in, const int* in_sizes, int n_in,
                              void* d_out, int out_size, void* d_ws, size_t ws_size,
                              hipStream_t stream) {
    (void)in_sizes; (void)n_in; (void)out_size; (void)d_ws; (void)ws_size;
    const float* emb  = (const float*)d_in[0];   // (B, 32, 64) f32
    const float* kern = (const float*)d_in[1];   // (64, 496, 64) f32
    float* out = (float*)d_out;                  // (B, 496) f32

    dim3 grid(PP, BB / 128);   // 496 pairs x 16 batch blocks
    opnl_wmma_kernel<<<grid, 256, 0, stream>>>(emb, kern, out);
}
